// MoELayer_52338471469501
// MI455X (gfx1250) — matrix operand order repaired and checked by reading, not compiled, not run on hardware
//
#include <hip/hip_runtime.h>
#include <hip/hip_bf16.h>

// ---------------------------------------------------------------------------
// MoE (top-2 of 16 experts), D = F = 1024, T = 16384 tokens.
// bf16 WMMA (v_wmma_f32_16x16x32_bf16) gather-based expert GEMM.
// M=32 token-row groups + 2-wide N blocking: 8 WMMAs per 12 b128 loads.
// ---------------------------------------------------------------------------

#define T_TOKENS   16384
#define DIM        1024
#define FFN        1024
#define NEXP       16
#define MAX_ROWS   (2 * T_TOKENS)          // exact sum of per-expert counts
#define MAX_PROWS  (MAX_ROWS + NEXP * 32)  // with per-expert padding to 32
#define MAX_GROUPS (MAX_ROWS / 32 + NEXP)  // 1040

typedef __bf16 bhalf_t;
typedef bhalf_t v8bf  __attribute__((ext_vector_type(8)));
typedef bhalf_t v16bf __attribute__((ext_vector_type(16)));
typedef float   v8f   __attribute__((ext_vector_type(8)));

// Build the 16-half WMMA operand from two 8-half (16B) loads.
__device__ __forceinline__ v16bf ld2x8(const bhalf_t* p0, const bhalf_t* p1) {
    v8bf lo = *(const v8bf*)p0;
    v8bf hi = *(const v8bf*)p1;
    return __builtin_shufflevector(lo, hi, 0,1,2,3,4,5,6,7,8,9,10,11,12,13,14,15);
}

__device__ __forceinline__ v8f wmma_bf16(v16bf a, v16bf b, v8f c) {
    return __builtin_amdgcn_wmma_f32_16x16x32_bf16(
        /*neg_a=*/false, a, /*neg_b=*/false, b,
        /*c_mod=*/(short)0, c, /*reuse_a=*/false, /*reuse_b=*/false);
}

__device__ __forceinline__ bhalf_t swiglu(float g, float u) {
    return (bhalf_t)((g / (1.f + __expf(-g))) * u);
}

// ---------------------------------------------------------------------------
// 1) fp32 -> bf16 conversion of token activations (coalesced, 4 elems/thread)
// ---------------------------------------------------------------------------
__global__ void moe_cvt_x(const float* __restrict__ in, bhalf_t* __restrict__ out, int n) {
    int i = (blockIdx.x * blockDim.x + threadIdx.x) * 4;
    if (i >= n) return;
    float4 v = *(const float4*)(in + i);
    out[i + 0] = (bhalf_t)v.x;
    out[i + 1] = (bhalf_t)v.y;
    out[i + 2] = (bhalf_t)v.z;
    out[i + 3] = (bhalf_t)v.w;
}

// ---------------------------------------------------------------------------
// 2) Transpose + convert one expert weight: in [E][R=1024][C=1024] fp32
//    -> out [E][C][R] bf16.  32x32 LDS tiles, block = (32,8).
// ---------------------------------------------------------------------------
__global__ void moe_transpose_cvt(const float* __restrict__ in, bhalf_t* __restrict__ out) {
    __shared__ float tile[32][33];
    const int e  = blockIdx.z;
    const int rb = blockIdx.y * 32;
    const int cb = blockIdx.x * 32;
    const int tx = threadIdx.x, ty = threadIdx.y;
    const float* src = in + ((size_t)e << 20);
    #pragma unroll
    for (int i = ty; i < 32; i += 8)
        tile[i][tx] = src[(size_t)(rb + i) * 1024 + cb + tx];
    __syncthreads();
    bhalf_t* dst = out + ((size_t)e << 20);
    #pragma unroll
    for (int i = ty; i < 32; i += 8)
        dst[(size_t)(cb + i) * 1024 + rb + tx] = (bhalf_t)tile[tx][i];
}

// ---------------------------------------------------------------------------
// 3) Router: one wave32 per token.  Computes logits (x @ router_w), top-2
//    softmax gate, per-expert counts, and softmax usage sums for the loss.
// ---------------------------------------------------------------------------
__global__ void moe_router(const float* __restrict__ x, const float* __restrict__ rw,
                           int* __restrict__ idx, float* __restrict__ wgt,
                           int* __restrict__ counts, float* __restrict__ usage) {
    const int lane = threadIdx.x & 31;
    const int wid  = threadIdx.x >> 5;
    const int t    = blockIdx.x * 8 + wid;

    __shared__ float susage[NEXP];
    if (threadIdx.x < NEXP) susage[threadIdx.x] = 0.f;
    __syncthreads();

    float acc[NEXP];
    #pragma unroll
    for (int e = 0; e < NEXP; e++) acc[e] = 0.f;

    const float* xr = x + (size_t)t * DIM;
    for (int i = 0; i < DIM / 32; i++) {
        const int d  = i * 32 + lane;
        const float xv = xr[d];
        const float4* r4 = (const float4*)(rw + (size_t)d * NEXP);
        float4 a = r4[0], b = r4[1], c = r4[2], q = r4[3];
        acc[0]  += xv * a.x;  acc[1]  += xv * a.y;  acc[2]  += xv * a.z;  acc[3]  += xv * a.w;
        acc[4]  += xv * b.x;  acc[5]  += xv * b.y;  acc[6]  += xv * b.z;  acc[7]  += xv * b.w;
        acc[8]  += xv * c.x;  acc[9]  += xv * c.y;  acc[10] += xv * c.z;  acc[11] += xv * c.w;
        acc[12] += xv * q.x;  acc[13] += xv * q.y;  acc[14] += xv * q.z;  acc[15] += xv * q.w;
    }
    #pragma unroll
    for (int off = 16; off > 0; off >>= 1) {
        #pragma unroll
        for (int e = 0; e < NEXP; e++) acc[e] += __shfl_xor(acc[e], off, 32);
    }

    if (lane == 0) {
        // top-2 (strict > keeps lowest index on ties, matching lax.top_k)
        float m0 = -3.0e38f; int i0 = 0;
        #pragma unroll
        for (int e = 0; e < NEXP; e++) if (acc[e] > m0) { m0 = acc[e]; i0 = e; }
        float m1 = -3.0e38f; int i1 = 0;
        #pragma unroll
        for (int e = 0; e < NEXP; e++) if (e != i0 && acc[e] > m1) { m1 = acc[e]; i1 = e; }
        const float e1 = __expf(m1 - m0);
        const float w0 = 1.f / (1.f + e1);
        idx[2 * t + 0] = i0;  wgt[2 * t + 0] = w0;
        idx[2 * t + 1] = i1;  wgt[2 * t + 1] = 1.f - w0;
        atomicAdd(&counts[i0], 1);
        atomicAdd(&counts[i1], 1);
        // full softmax for expert_usage / load-balancing loss
        float s = 0.f, ex[NEXP];
        #pragma unroll
        for (int e = 0; e < NEXP; e++) { ex[e] = __expf(acc[e] - m0); s += ex[e]; }
        const float inv = 1.f / s;
        #pragma unroll
        for (int e = 0; e < NEXP; e++) atomicAdd(&susage[e], ex[e] * inv);
    }
    __syncthreads();
    if (threadIdx.x < NEXP) unsafeAtomicAdd(&usage[threadIdx.x], susage[threadIdx.x]);
}

// ---------------------------------------------------------------------------
// 4) Finalize: padded offsets (rows to 32), cumulative group counts, loss.
// ---------------------------------------------------------------------------
__global__ void moe_finalize(const int* __restrict__ counts, const float* __restrict__ usage,
                             int* __restrict__ poff, int* __restrict__ cg,
                             float* __restrict__ loss_out) {
    if (threadIdx.x != 0 || blockIdx.x != 0) return;
    int po = 0, c = 0;
    #pragma unroll
    for (int e = 0; e < NEXP; e++) {
        poff[e] = po; cg[e] = c;
        const int g = (counts[e] + 31) >> 5;
        c += g; po += g * 32;
    }
    poff[NEXP] = po; cg[NEXP] = c;
    float s = 0.f;
    #pragma unroll
    for (int e = 0; e < NEXP; e++) {
        const float u = usage[e] * (1.f / (float)T_TOKENS);
        s += u * u;
    }
    *loss_out = (float)NEXP * s;
}

// ---------------------------------------------------------------------------
// 5) Scatter tokens into padded per-expert row lists (padding pre-zeroed:
//    row 0 / gate 0 -> contributes exactly 0 to the output).
// ---------------------------------------------------------------------------
__global__ void moe_scatter(const int* __restrict__ idx, const float* __restrict__ wgt,
                            const int* __restrict__ poff, int* __restrict__ fill,
                            int* __restrict__ rows, float* __restrict__ gates) {
    const int t = blockIdx.x * blockDim.x + threadIdx.x;
    if (t >= T_TOKENS) return;
    #pragma unroll
    for (int k = 0; k < 2; k++) {
        const int e = idx[2 * t + k];
        const float w = wgt[2 * t + k];
        const int p = atomicAdd(&fill[e], 1);
        rows[poff[e] + p]  = t;
        gates[poff[e] + p] = w;
    }
}

// ---------------------------------------------------------------------------
// 6) Expert GEMM.  One block = (expert e, group of 32 token-rows).
//    256 threads = 8 waves; each wave owns 32-wide N strips (two WMMA tiles).
//    Phase 1: h = silu(x@w1) * (x@w3)  (bf16 WMMA, f32 acc) -> 64KB LDS bf16
//             8 WMMAs per 12 global_load_b128 per K-step (A/B fully reused).
//    Phase 2: out += gate * (h @ w2)   (A from LDS via ds_load_b128) -> atomics
//
//  WMMA operand K order used here (wave32, 16-bit; contiguous, not AMD's table order):
//    A 16x32 : lane<16 -> M=lane,    halves = K {k0..k0+15}
//              lane>=16-> M=lane-16, halves = K {k0+16..k0+31}
//    B 32x16 : lane<16 -> N=lane,    halves = K {k0..k0+15}
//              lane>=16-> N=lane-16, halves = K {k0+16..k0+31}
//    C/D f32 : vgpr r -> M = r (+8 for lane>=16), N = lane&15
// ---------------------------------------------------------------------------
__global__ void __launch_bounds__(256)
moe_expert(const bhalf_t* __restrict__ xbf,
           const bhalf_t* __restrict__ w1t,   // [E][F][D]
           const bhalf_t* __restrict__ w3t,   // [E][F][D]
           const bhalf_t* __restrict__ w2t,   // [E][D][F]
           const int* __restrict__ rows, const float* __restrict__ gates,
           const int* __restrict__ poff, const int* __restrict__ cg,
           float* __restrict__ out) {
    __shared__ bhalf_t hlds[32 * FFN];   // exactly 64 KB SwiGLU activations

    const int b = blockIdx.x;
    if (b >= cg[NEXP]) return;
    int e = 0;
    while (e < NEXP - 1 && b >= cg[e + 1]) e++;
    const int grp  = b - cg[e];
    const int base = poff[e] + grp * 32;

    const int lane   = threadIdx.x & 31;
    const int wid    = threadIdx.x >> 5;
    const int mlane  = lane & 15;
    const bool hiH   = lane >= 16;
    const int aoff0  = hiH ? 16 : 0;    // A-operand K sub-offsets
    const int aoff1  = hiH ? 24 : 8;
    const int bkoff  = hiH ? 16 : 0;    // B-operand K sub-offset

    // Lane-held row ids / gates (lane m holds entry m of each 16-row tile).
    const int   row_lo = rows [base + mlane];
    const int   row_hi = rows [base + 16 + mlane];
    const float gat_lo = gates[base + mlane];
    const float gat_hi = gates[base + 16 + mlane];
    const bhalf_t* arow0 = xbf + (size_t)row_lo * DIM;
    const bhalf_t* arow1 = xbf + (size_t)row_hi * DIM;

    // ---------------- Phase 1: h = silu(x@w1) * (x@w3) --------------------
    for (int ncp = wid; ncp < FFN / 32; ncp += 8) {
        const int n0 = ncp * 32;
        const bhalf_t* b1p0 = w1t + ((size_t)e * FFN + n0 + mlane) * DIM + bkoff;
        const bhalf_t* b1p1 = b1p0 + (size_t)16 * DIM;
        const bhalf_t* b3p0 = w3t + ((size_t)e * FFN + n0 + mlane) * DIM + bkoff;
        const bhalf_t* b3p1 = b3p0 + (size_t)16 * DIM;
        v8f g00 = {0,0,0,0,0,0,0,0}, g01 = {0,0,0,0,0,0,0,0};
        v8f g10 = {0,0,0,0,0,0,0,0}, g11 = {0,0,0,0,0,0,0,0};
        v8f u00 = {0,0,0,0,0,0,0,0}, u01 = {0,0,0,0,0,0,0,0};
        v8f u10 = {0,0,0,0,0,0,0,0}, u11 = {0,0,0,0,0,0,0,0};
        #pragma unroll 2
        for (int k0 = 0; k0 < DIM; k0 += 32) {
            v16bf a0  = ld2x8(arow0 + k0 + aoff0, arow0 + k0 + aoff1);
            v16bf a1  = ld2x8(arow1 + k0 + aoff0, arow1 + k0 + aoff1);
            v16bf b10 = ld2x8(b1p0 + k0, b1p0 + k0 + 8);
            v16bf b11 = ld2x8(b1p1 + k0, b1p1 + k0 + 8);
            v16bf b30 = ld2x8(b3p0 + k0, b3p0 + k0 + 8);
            v16bf b31 = ld2x8(b3p1 + k0, b3p1 + k0 + 8);
            g00 = wmma_bf16(a0, b10, g00);  g01 = wmma_bf16(a0, b11, g01);
            g10 = wmma_bf16(a1, b10, g10);  g11 = wmma_bf16(a1, b11, g11);
            u00 = wmma_bf16(a0, b30, u00);  u01 = wmma_bf16(a0, b31, u01);
            u10 = wmma_bf16(a1, b30, u10);  u11 = wmma_bf16(a1, b31, u11);
        }
        #pragma unroll
        for (int r = 0; r < 8; r++) {
            const int m0 = r + (hiH ? 8 : 0);
            const int m1 = m0 + 16;
            hlds[m0 * FFN + n0 + mlane]      = swiglu(g00[r], u00[r]);
            hlds[m0 * FFN + n0 + 16 + mlane] = swiglu(g01[r], u01[r]);
            hlds[m1 * FFN + n0 + mlane]      = swiglu(g10[r], u10[r]);
            hlds[m1 * FFN + n0 + 16 + mlane] = swiglu(g11[r], u11[r]);
        }
    }
    __syncthreads();

    // ---------------- Phase 2: out += gate * (h @ w2) ---------------------
    const bhalf_t* ap0 = hlds + (size_t)mlane * FFN;
    const bhalf_t* ap1 = hlds + (size_t)(mlane + 16) * FFN;
    for (int ncp = wid; ncp < DIM / 32; ncp += 8) {
        const int n0 = ncp * 32;
        const bhalf_t* b2p0 = w2t + ((size_t)e * DIM + n0 + mlane) * FFN + bkoff;
        const bhalf_t* b2p1 = b2p0 + (size_t)16 * FFN;
        v8f c00 = {0,0,0,0,0,0,0,0}, c01 = {0,0,0,0,0,0,0,0};
        v8f c10 = {0,0,0,0,0,0,0,0}, c11 = {0,0,0,0,0,0,0,0};
        #pragma unroll 2
        for (int k0 = 0; k0 < FFN; k0 += 32) {
            v16bf a0 = ld2x8(ap0 + k0 + aoff0, ap0 + k0 + aoff1);   // ds_load_b128
            v16bf a1 = ld2x8(ap1 + k0 + aoff0, ap1 + k0 + aoff1);
            v16bf b0 = ld2x8(b2p0 + k0, b2p0 + k0 + 8);
            v16bf b1 = ld2x8(b2p1 + k0, b2p1 + k0 + 8);
            c00 = wmma_bf16(a0, b0, c00);  c01 = wmma_bf16(a0, b1, c01);
            c10 = wmma_bf16(a1, b0, c10);  c11 = wmma_bf16(a1, b1, c11);
        }
        #pragma unroll
        for (int r = 0; r < 8; r++) {
            const int m = r + (hiH ? 8 : 0);
            const int   rowA = __shfl(row_lo, m, 32);   // lane m holds row m
            const float gA   = __shfl(gat_lo, m, 32);
            const int   rowB = __shfl(row_hi, m, 32);
            const float gB   = __shfl(gat_hi, m, 32);
            unsafeAtomicAdd(&out[(size_t)rowA * DIM + n0 + mlane],      c00[r] * gA);
            unsafeAtomicAdd(&out[(size_t)rowA * DIM + n0 + 16 + mlane], c01[r] * gA);
            unsafeAtomicAdd(&out[(size_t)rowB * DIM + n0 + mlane],      c10[r] * gB);
            unsafeAtomicAdd(&out[(size_t)rowB * DIM + n0 + 16 + mlane], c11[r] * gB);
        }
    }
}

// ---------------------------------------------------------------------------
// Host-side orchestration (graph-capture safe: only async ops on `stream`).
// Workspace layout:
//   [0, ~520KB) control (counts/fill/usage/poff/cg/idx/wgt/rows/gates) zeroed
//   [1MB)     xbf   : T*D bf16          (32 MB)
//   [+32MB)   w1t   : E*F*D bf16        (32 MB)
//   [+64MB)   w3t   : E*F*D bf16        (32 MB)
//   [+96MB)   w2t   : E*D*F bf16        (32 MB)      total ~129 MB
// ---------------------------------------------------------------------------
extern "C" void kernel_launch(void* const* d_in, const int* in_sizes, int n_in,
                              void* d_out, int out_size, void* d_ws, size_t ws_size,
                              hipStream_t stream) {
    (void)in_sizes; (void)n_in; (void)ws_size;
    const float* x  = (const float*)d_in[0];
    const float* rw = (const float*)d_in[1];
    const float* w1 = (const float*)d_in[2];
    const float* w2 = (const float*)d_in[3];
    const float* w3 = (const float*)d_in[4];
    float* out = (float*)d_out;

    char* ws = (char*)d_ws;
    // --- control region ---
    int*   counts = (int*)  (ws + 0);
    int*   fill   = (int*)  (ws + 64);
    float* usage  = (float*)(ws + 128);
    int*   poff   = (int*)  (ws + 256);
    int*   cg     = (int*)  (ws + 384);
    int*   idx    = (int*)  (ws + 1024);
    float* wgt    = (float*)(ws + 1024 + 2 * T_TOKENS * sizeof(int));
    int*   rows   = (int*)  (ws + 1024 + 4 * T_TOKENS * sizeof(int));
    float* gates  = (float*)(ws + 1024 + 4 * T_TOKENS * sizeof(int) + MAX_PROWS * sizeof(int));
    const size_t ctrl_bytes =
        1024 + 4 * (size_t)T_TOKENS * sizeof(int) + 2 * (size_t)MAX_PROWS * sizeof(int);
    // --- bf16 region ---
    const size_t MB = 1u << 20;
    bhalf_t* xbf = (bhalf_t*)(ws + 1 * MB);
    bhalf_t* w1t = (bhalf_t*)(ws + 33 * MB);
    bhalf_t* w3t = (bhalf_t*)(ws + 65 * MB);
    bhalf_t* w2t = (bhalf_t*)(ws + 97 * MB);

    hipMemsetAsync(out, 0, (size_t)out_size * sizeof(float), stream);
    hipMemsetAsync(ws, 0, ctrl_bytes, stream);

    // bf16 conversions (weights become L2-resident: 96 MB < 192 MB L2)
    moe_cvt_x<<<(T_TOKENS * DIM / 4 + 255) / 256, 256, 0, stream>>>(x, xbf, T_TOKENS * DIM);
    dim3 tg(32, 32, NEXP), tb(32, 8);
    moe_transpose_cvt<<<tg, tb, 0, stream>>>(w1, w1t);
    moe_transpose_cvt<<<tg, tb, 0, stream>>>(w3, w3t);
    moe_transpose_cvt<<<tg, tb, 0, stream>>>(w2, w2t);

    // routing + bookkeeping
    moe_router<<<T_TOKENS / 8, 256, 0, stream>>>(x, rw, idx, wgt, counts, usage);
    moe_finalize<<<1, 32, 0, stream>>>(counts, usage, poff, cg, out + (out_size - 1));
    moe_scatter<<<T_TOKENS / 256, 256, 0, stream>>>(idx, wgt, poff, fill, rows, gates);

    // expert SwiGLU GEMMs (bf16 WMMA, 32-row groups)
    moe_expert<<<MAX_GROUPS, 256, 0, stream>>>(xbf, w1t, w3t, w2t,
                                               rows, gates, poff, cg, out);
}
